// MultiHeadAttention_62105227100510
// MI455X (gfx1250) — compile-verified, matrix operand order checked by reading, not run on hardware
//
#include <hip/hip_runtime.h>
#include <hip/hip_bf16.h>
#include <stdint.h>

// ---------------------------------------------------------------- types
typedef __bf16 bf16;
typedef __attribute__((ext_vector_type(16))) __bf16 v16bf;
typedef __attribute__((ext_vector_type(8)))  __bf16 v8bf;
typedef __attribute__((ext_vector_type(8)))  float  v8f;

#define BB   4
#define SS   1024
#define HH   12
#define DKK  64
#define DVV  64
#define DMM  768
#define NEGBIG (-1e12f)

// ---------------------------------------------------------------- helpers
__device__ __forceinline__ unsigned lane_id() { return threadIdx.x & 31u; }

__device__ __forceinline__ bf16 f2bf(float f) {
  union { float f; uint32_t u; } a; a.f = f;
  uint32_t r = (a.u + 0x7fffu + ((a.u >> 16) & 1u)) >> 16;   // RNE
  union { uint16_t s; bf16 b; } o; o.s = (uint16_t)r; return o.b;
}
__device__ __forceinline__ float bf2f(bf16 b) {
  union { uint16_t s; bf16 b; } i; i.b = b;
  union { uint32_t u; float f; } o; o.u = ((uint32_t)i.s) << 16; return o.f;
}

// CDNA5 async global->LDS copy (16B per lane), tracked by ASYNCcnt.
__device__ __forceinline__ void async_g2l_b128(uint32_t lds_off, const void* g) {
  asm volatile("global_load_async_to_lds_b128 %0, %1, off"
               :: "v"(lds_off), "v"(g) : "memory");
}
__device__ __forceinline__ void wait_async0() {
  asm volatile("s_wait_asynccnt 0x0" ::: "memory");
}

// Load one 16x32 bf16 A-fragment (or 32x16 B-fragment from a transposed,
// row-per-N buffer). ISA 7.12.2 layout:
//   lane L (0..15):  row = L,      K = {0..7, 16..23}
//   lane L+16:       row = L,      K = {8..15, 24..31}
__device__ __forceinline__ v16bf load_frag(const bf16* base, int stride) {
  const unsigned l   = lane_id();
  const unsigned r   = l & 15u;
  const unsigned klo = (l < 16u) ? 0u : 8u;
  const bf16* p = base + (size_t)r * (size_t)stride + klo;
  v8bf lo = *(const v8bf*)(p);
  v8bf hi = *(const v8bf*)(p + 16);
  return __builtin_shufflevector(lo, hi, 0,1,2,3,4,5,6,7,8,9,10,11,12,13,14,15);
}

__device__ __forceinline__ v8f wmma_bf16(v16bf a, v16bf b, v8f c) {
  return __builtin_amdgcn_wmma_f32_16x16x32_bf16(false, a, false, b,
                                                 (short)0, c, false, false);
}

// ---------------------------------------------------------------- GEMM
// C[M,N] = A[M,K] @ B[K,N] + bias[N]; row-major; block tile 64x64,
// 4 waves x (16 rows x 64 cols), K-step 32, bf16 WMMA w/ f32 accum.
template <bool A_F32, bool B_F32, bool OUT_F32>
__global__ void __launch_bounds__(128, 4)
gemm_rr(const void* __restrict__ Ap, const void* __restrict__ Bp,
        const float* __restrict__ bias, void* __restrict__ Cp,
        int M, int N, int K)
{
  __shared__ bf16 As[64][40];   // rows: M-tile, cols: k (pad 40 -> 80B stride)
  __shared__ bf16 Bs[64][40];   // rows: N-tile, cols: k (transposed stage)

  const int tid  = threadIdx.x;
  const int wave = tid >> 5;
  const unsigned l  = tid & 31u;
  const unsigned kl = l & 15u;
  const int half = (l < 16u) ? 0 : 8;
  const int m0 = blockIdx.x * 64;
  const int n0 = blockIdx.y * 64;

  v8f acc[4] = {};

  for (int k0 = 0; k0 < K; k0 += 32) {
    __syncthreads();
    // ---- stage A tile (64x32)
    for (int c = tid; c < 512; c += 128) {
      int r = c >> 3, c4 = (c & 7) * 4;
      if constexpr (A_F32) {
        const float* A = (const float*)Ap;
        const float* s = A + (size_t)(m0 + r) * K + k0 + c4;
        bf16* d = &As[r][c4];
        d[0] = f2bf(s[0]); d[1] = f2bf(s[1]); d[2] = f2bf(s[2]); d[3] = f2bf(s[3]);
      } else {
        const bf16* A = (const bf16*)Ap;
        *(uint64_t*)&As[r][c4] =
            *(const uint64_t*)(A + (size_t)(m0 + r) * K + k0 + c4);
      }
    }
    // ---- stage B tile (32x64) transposed into Bs[n][k]
    for (int c = tid; c < 512; c += 128) {
      int kk = c >> 4, n4 = (c & 15) * 4;
      if constexpr (B_F32) {
        const float* Bm = (const float*)Bp;
        const float* s = Bm + (size_t)(k0 + kk) * N + n0 + n4;
        Bs[n4 + 0][kk] = f2bf(s[0]); Bs[n4 + 1][kk] = f2bf(s[1]);
        Bs[n4 + 2][kk] = f2bf(s[2]); Bs[n4 + 3][kk] = f2bf(s[3]);
      } else {
        const bf16* Bm = (const bf16*)Bp;
        const bf16* s = Bm + (size_t)(k0 + kk) * N + n0 + n4;
        Bs[n4 + 0][kk] = s[0]; Bs[n4 + 1][kk] = s[1];
        Bs[n4 + 2][kk] = s[2]; Bs[n4 + 3][kk] = s[3];
      }
    }
    __syncthreads();

    v16bf av = load_frag(&As[wave * 16][0], 40);
#pragma unroll
    for (int ns = 0; ns < 4; ++ns) {
      v16bf bv = load_frag(&Bs[ns * 16][0], 40);
      acc[ns] = wmma_bf16(av, bv, acc[ns]);
    }
  }

  // ---- epilogue (C layout: lane<16 -> M=i, lane>=16 -> M=8+i; N = lane&15)
#pragma unroll
  for (int ns = 0; ns < 4; ++ns) {
    int col = n0 + ns * 16 + (int)kl;
    float bv = bias ? bias[col] : 0.f;
#pragma unroll
    for (int i = 0; i < 8; ++i) {
      int row = m0 + wave * 16 + i + half;
      float v = acc[ns][i] + bv;
      if constexpr (OUT_F32)
        ((float*)Cp)[(size_t)row * N + col] = v;
      else
        ((bf16*)Cp)[(size_t)row * N + col] = f2bf(v);
    }
  }
}

// ---------------------------------------------------------------- attention
// grid = (B*H fast, S/64), block = 128 (4 waves x 16 j-rows each).
// Flash-style online softmax; WMMA for QK^T and P@V; VALU fp32 for the two
// position-bias einsums; pb kept fp32 in global (L2-resident by grid order).
// K tile staged with CDNA5 async global->LDS; V staged+transposed in VALU.
// Single accumulator: VALU P@pb contributions fold into the WMMA accumulator
// (both are rescaled by the same online-softmax correction factor).
__global__ void __launch_bounds__(128, 4)
attn_kernel(const bf16* __restrict__ qw, const bf16* __restrict__ kw,
            const bf16* __restrict__ vw, const float* __restrict__ pb,
            const float* __restrict__ amask, const int* __restrict__ vmask,
            bf16* __restrict__ attn)
{
  __shared__ bf16  Kt[64][72];       // [key][d]
  __shared__ bf16  Vt[64][72];       // transposed: [d][key]
  __shared__ float qf[64][65];       // fp32 q rows for pb dot
  __shared__ bf16  Pt[4][16][72];    // per-wave P tile [wave][j][k]

  const int b  = blockIdx.x / HH;
  const int h  = blockIdx.x % HH;
  const int j0 = blockIdx.y * 64;
  const int tid = threadIdx.x, wave = tid >> 5;
  const unsigned l  = tid & 31u;
  const unsigned kl = l & 15u;
  const int half = (l < 16u) ? 0 : 8;

  // stage fp32 q rows (from bf16 qw) for the VALU pb dot
  for (int c = tid; c < 64 * 16; c += 128) {
    int r = c >> 4, d4 = (c & 15) * 4;
    const bf16* s = qw + (((size_t)(b * SS + j0 + r) * HH) + h) * DKK + d4;
    uint64_t raw = *(const uint64_t*)s;
    bf16 t4[4]; *(uint64_t*)t4 = raw;
    qf[r][d4 + 0] = bf2f(t4[0]); qf[r][d4 + 1] = bf2f(t4[1]);
    qf[r][d4 + 2] = bf2f(t4[2]); qf[r][d4 + 3] = bf2f(t4[3]);
  }

  // per-wave Q fragments straight from global (rows stride H*DK)
  const bf16* qbase = qw + (((size_t)(b * SS + j0 + wave * 16) * HH) + h) * DKK;
  v16bf qa0 = load_frag(qbase,      HH * DKK);   // d 0..31
  v16bf qa1 = load_frag(qbase + 32, HH * DKK);   // d 32..63

  v8f accv[4] = {};   // combined accumulator: P@V (WMMA) + P@pb (VALU)
  float rowm[8], rowl[8];
#pragma unroll
  for (int i = 0; i < 8; ++i) { rowm[i] = -3.0e38f; rowl[i] = 0.f; }

  for (int kt = 0; kt < SS; kt += 64) {
    __syncthreads();
    // ---- stage K tile via async global->LDS (8 bf16 = 16B per issue)
#pragma unroll
    for (int c = tid; c < 64 * 8; c += 128) {
      int r = c >> 3, d8 = (c & 7) * 8;
      async_g2l_b128((uint32_t)(size_t)&Kt[r][d8],
                     kw + (((size_t)(b * SS + kt + r) * HH) + h) * DKK + d8);
    }
    // ---- stage V tile transposed (overlaps with the async K copy)
    for (int c = tid; c < 64 * 16; c += 128) {
      int r = c >> 4, d4 = (c & 15) * 4;
      uint64_t raw = *(const uint64_t*)
          (vw + (((size_t)(b * SS + kt + r) * HH) + h) * DKK + d4);
      bf16 t4[4]; *(uint64_t*)t4 = raw;
      Vt[d4 + 0][r] = t4[0]; Vt[d4 + 1][r] = t4[1];
      Vt[d4 + 2][r] = t4[2]; Vt[d4 + 3][r] = t4[3];
    }
    // ---- prefetch next pb slab into L2 while we compute this tile
    if (kt + 64 < SS) {
      const float* nf = pb +
          ((size_t)(j0 + (tid >> 1)) * SS + kt + 64 + (tid & 1) * 32) * DKK;
      __builtin_prefetch(nf, 0, 1);
    }
    wait_async0();
    __syncthreads();

    // ---- QK^T: 4 key-subtiles of 16, K accumulated over d=64 (2 WMMA each)
    v8f sc[4];
#pragma unroll
    for (int kc = 0; kc < 4; ++kc) {
      v16bf kb0 = load_frag(&Kt[kc * 16][0],  72);
      v16bf kb1 = load_frag(&Kt[kc * 16][32], 72);
      v8f z = {};
      z = wmma_bf16(qa0, kb0, z);
      sc[kc] = wmma_bf16(qa1, kb1, z);
    }

    // ---- position-bias score term + scale + masks (VALU fp32)
#pragma unroll
    for (int i = 0; i < 8; ++i) {
      const int jl = wave * 16 + i + half;
      const int jg = j0 + jl;
      const float* qrow = &qf[jl][0];
      float dot[4] = {0.f, 0.f, 0.f, 0.f};
      const float* pbp[4];
#pragma unroll
      for (int kc = 0; kc < 4; ++kc)
        pbp[kc] = pb + ((size_t)jg * SS + kt + kc * 16 + kl) * DKK;
      for (int dd = 0; dd < 64; dd += 4) {
        float q0 = qrow[dd], q1 = qrow[dd + 1], q2 = qrow[dd + 2], q3 = qrow[dd + 3];
#pragma unroll
        for (int kc = 0; kc < 4; ++kc) {
          const float* p = pbp[kc] + dd;
          dot[kc] += q0 * p[0] + q1 * p[1] + q2 * p[2] + q3 * p[3];
        }
      }
#pragma unroll
      for (int kc = 0; kc < 4; ++kc) {
        int kg = kt + kc * 16 + (int)kl;
        float s = (sc[kc][i] + dot[kc]) * 0.125f + amask[(size_t)jg * SS + kg];
        s = vmask[b * SS + kg] ? s : NEGBIG;
        sc[kc][i] = s;
      }
    }

    // ---- online softmax (row stats live per 16-lane half, xor-reduced)
#pragma unroll
    for (int i = 0; i < 8; ++i) {
      float lm = fmaxf(fmaxf(sc[0][i], sc[1][i]), fmaxf(sc[2][i], sc[3][i]));
      lm = fmaxf(lm, __shfl_xor(lm, 1));
      lm = fmaxf(lm, __shfl_xor(lm, 2));
      lm = fmaxf(lm, __shfl_xor(lm, 4));
      lm = fmaxf(lm, __shfl_xor(lm, 8));
      float mnew = fmaxf(rowm[i], lm);
      float corr = __expf(rowm[i] - mnew);
      rowm[i] = mnew;
      float ps = 0.f;
#pragma unroll
      for (int kc = 0; kc < 4; ++kc) {
        float p = __expf(sc[kc][i] - mnew);
        sc[kc][i] = p;
        ps += p;
      }
      ps += __shfl_xor(ps, 1);
      ps += __shfl_xor(ps, 2);
      ps += __shfl_xor(ps, 4);
      ps += __shfl_xor(ps, 8);
      rowl[i] = rowl[i] * corr + ps;
#pragma unroll
      for (int dc = 0; dc < 4; ++dc) accv[dc][i] *= corr;
    }

    // ---- P -> LDS (bf16), C-layout to A-fragment layout round trip
#pragma unroll
    for (int i = 0; i < 8; ++i)
#pragma unroll
      for (int kc = 0; kc < 4; ++kc)
        Pt[wave][i + half][kc * 16 + kl] = f2bf(sc[kc][i]);

    // ---- P @ V (WMMA): 4 d-subtiles, K over 64 keys (2 WMMA each)
    v16bf pa0 = load_frag(&Pt[wave][0][0],  72);
    v16bf pa1 = load_frag(&Pt[wave][0][32], 72);
#pragma unroll
    for (int dc = 0; dc < 4; ++dc) {
      v16bf vb0 = load_frag(&Vt[dc * 16][0],  72);
      v16bf vb1 = load_frag(&Vt[dc * 16][32], 72);
      accv[dc] = wmma_bf16(pa0, vb0, accv[dc]);
      accv[dc] = wmma_bf16(pa1, vb1, accv[dc]);
    }

    // ---- P @ pb (VALU fp32) folded into the same accumulator
#pragma unroll
    for (int i = 0; i < 8; ++i) {
      const size_t base = ((size_t)(j0 + wave * 16 + i + half) * SS + kt) * DKK + kl;
      const bf16* prow = &Pt[wave][i + half][0];
      for (int kk = 0; kk < 64; kk += 4) {
        uint64_t raw = *(const uint64_t*)(prow + kk);   // 4 probs, one ds_load_b64
        bf16 p4[4]; *(uint64_t*)p4 = raw;
#pragma unroll
        for (int u = 0; u < 4; ++u) {
          float p = bf2f(p4[u]);
          const float* pr = pb + base + (size_t)(kk + u) * DKK;
          accv[0][i] += p * pr[0];
          accv[1][i] += p * pr[16];
          accv[2][i] += p * pr[32];
          accv[3][i] += p * pr[48];
        }
      }
    }
  }

  // ---- finalize: accv / rowsum -> attn[b, j, h, d] (bf16)
#pragma unroll
  for (int i = 0; i < 8; ++i) {
    const int jg = j0 + wave * 16 + i + half;
    float inv = 1.0f / rowl[i];
    bf16* dst = attn + (((size_t)(b * SS + jg) * HH) + h) * DVV + kl;
#pragma unroll
    for (int dc = 0; dc < 4; ++dc)
      dst[dc * 16] = f2bf(accv[dc][i] * inv);
  }
}

// ---------------------------------------------------------------- launcher
extern "C" void kernel_launch(void* const* d_in, const int* in_sizes, int n_in,
                              void* d_out, int out_size, void* d_ws, size_t ws_size,
                              hipStream_t stream)
{
  (void)in_sizes; (void)n_in; (void)out_size; (void)ws_size;
  const float* q     = (const float*)d_in[0];
  const float* k     = (const float*)d_in[1];
  const float* v     = (const float*)d_in[2];
  const float* amask = (const float*)d_in[3];
  const float* pb    = (const float*)d_in[4];
  const int*   vmask = (const int*)d_in[5];
  const float* Wq    = (const float*)d_in[6];
  const float* bq    = (const float*)d_in[7];
  const float* Wk    = (const float*)d_in[8];
  const float* bk    = (const float*)d_in[9];
  const float* Wv    = (const float*)d_in[10];
  const float* bv    = (const float*)d_in[11];
  const float* Wo    = (const float*)d_in[12];
  const float* bo    = (const float*)d_in[13];

  const size_t NE = (size_t)BB * SS * DMM;     // 3,145,728 elements
  bf16* qw   = (bf16*)d_ws;                    // bf16 activations in workspace
  bf16* kwp  = qw  + NE;
  bf16* vwp  = kwp + NE;
  bf16* attn = vwp + NE;                       // total 4*NE*2 = 25.2 MB

  const int M = BB * SS;                       // 4096
  const int N = DMM;                           // 768
  const int K = DMM;                           // 768
  dim3 blk(128);
  dim3 gg(M / 64, N / 64);                     // 64 x 12

  // projections: fp32 in -> bf16 out, bias fused
  gemm_rr<true, true, false><<<gg, blk, 0, stream>>>(q, Wq, bq, qw,  M, N, K);
  gemm_rr<true, true, false><<<gg, blk, 0, stream>>>(k, Wk, bk, kwp, M, N, K);
  gemm_rr<true, true, false><<<gg, blk, 0, stream>>>(v, Wv, bv, vwp, M, N, K);

  // attention: (b,h) fast in grid.x so blocks sharing a pb j-slab co-schedule
  attn_kernel<<<dim3(BB * HH, SS / 64), blk, 0, stream>>>(
      qw, kwp, vwp, pb, amask, vmask, attn);

  // output projection: bf16 A, fp32 W/out
  gemm_rr<false, true, true><<<gg, blk, 0, stream>>>(attn, Wo, bo, d_out, M, N, K);
}
